// ContentBasedAttention_34205119546002
// MI455X (gfx1250) — matrix operand order repaired and checked by reading, not compiled, not run on hardware
//
#include <hip/hip_runtime.h>
#include <hip/hip_bf16.h>
#include <math.h>

typedef _Float16 f16;
typedef __attribute__((ext_vector_type(16))) _Float16 v16h;
typedef __attribute__((ext_vector_type(8)))  _Float16 v8h;
typedef __attribute__((ext_vector_type(8)))  float    v8f;

#define Hh   320
#define Eh   640
#define Bb   32
#define Tt   2000
#define Kk   100
#define NFk  10
#define PADc 50
#define KAUG 672   // 640 (h) + 10 (f) padded to 21*32
#define KLDS 680   // LDS row stride (f16) to de-conflict banks

// ---------------- prep: s_comb = s@W_se + b_se + b_he + b_fe ----------------
__global__ void k_scomb(const float* __restrict__ s, const float* __restrict__ W_se,
                        const float* __restrict__ b_se, const float* __restrict__ b_he,
                        const float* __restrict__ b_fe, float* __restrict__ scomb) {
  int idx = blockIdx.x * blockDim.x + threadIdx.x;
  if (idx >= Bb * Eh) return;
  int b = idx / Eh, e = idx % Eh;
  float acc = b_se[e] + b_he[e] + b_fe[e];
  const float* sp = s + b * Hh;
  for (int h = 0; h < Hh; ++h) acc = fmaf(sp[h], W_se[h * Eh + e], acc);
  scomb[idx] = acc;
}

// ------- prep: combined transposed f16 weights WgT[e][k], k<640 from W_he,
//         k in [640,650) from W_fe, rest zero -------
__global__ void k_wgt(const float* __restrict__ W_he, const float* __restrict__ W_fe,
                      f16* __restrict__ WgT) {
  int idx = blockIdx.x * blockDim.x + threadIdx.x;
  if (idx >= Eh * KAUG) return;
  int e = idx / KAUG, k = idx % KAUG;
  float v = 0.f;
  if (k < Eh) v = W_he[k * Eh + e];
  else if (k < Eh + NFk) v = W_fe[(k - Eh) * Eh + e];
  WgT[idx] = (f16)v;
}

// ---------------- prep: location conv, f[b][t][j] ----------------
__global__ __launch_bounds__(256) void k_conv(const float* __restrict__ alpha,
                                              const float* __restrict__ conv_w,
                                              float* __restrict__ f_ws) {
  __shared__ float w[NFk * Kk];
  for (int i = threadIdx.x; i < NFk * Kk; i += 256) w[i] = conv_w[i];
  __syncthreads();
  int idx = blockIdx.x * blockDim.x + threadIdx.x;
  if (idx >= Bb * Tt) return;
  int b = idx / Tt, t = idx % Tt;
  float acc[NFk];
#pragma unroll
  for (int j = 0; j < NFk; ++j) acc[j] = 0.f;
  const float* ap = alpha + b * Tt;
  for (int k = 0; k < Kk; ++k) {
    int ti = t + k - PADc;
    if (ti >= 0 && ti < Tt) {
      float a = ap[ti];
#pragma unroll
      for (int j = 0; j < NFk; ++j) acc[j] = fmaf(a, w[j * Kk + k], acc[j]);
    }
  }
#pragma unroll
  for (int j = 0; j < NFk; ++j) f_ws[idx * NFk + j] = acc[j];
}

// ---------------- main fused energy kernel (WMMA f16->f32) ----------------
// block = 256 threads (8 waves); one block per (b, 16-row tile of t).
// wave w covers e-columns [80w, 80w+80) as 5 WMMA accumulators.
__global__ __launch_bounds__(256) void k_energy(
    const float* __restrict__ h, const float* __restrict__ f_ws,
    const float* __restrict__ scomb, const f16* __restrict__ WgT,
    const float* __restrict__ W_ee, const float* __restrict__ b_ee,
    const unsigned char* __restrict__ mask, float* __restrict__ e_ws) {
  __shared__ __align__(32) f16 Ahat[16 * KLDS];
  __shared__ float part[8][16];

  int b  = blockIdx.x / (Tt / 16);
  int t0 = (blockIdx.x % (Tt / 16)) * 16;
  int tid = threadIdx.x;

  // Stage A tile: rows = 16 t's, cols = 640 h (f32->f16) + 10 f + zero pad
  const float* hb = h + ((size_t)b * Tt + t0) * Eh;
  for (int i = tid; i < 16 * Eh; i += 256) {
    int r = i / Eh, k = i % Eh;
    Ahat[r * KLDS + k] = (f16)hb[(size_t)r * Eh + k];
  }
  const float* fb = f_ws + ((size_t)b * Tt + t0) * NFk;
  for (int i = tid; i < 16 * 32; i += 256) {
    int r = i / 32, k = Eh + (i % 32);
    float v = (k < Eh + NFk) ? fb[r * NFk + (k - Eh)] : 0.f;
    Ahat[r * KLDS + k] = (f16)v;
  }
  __syncthreads();

  int w = tid >> 5, lane = tid & 31;
  int half = lane >> 4, m = lane & 15;
  int ebase = w * 80;

  v8f acc[5] = {};
  // A frag: lanes 0-15 hold M=lane, K {k0..k0+15};
  //         lanes 16-31 hold K {k0+16..k0+31}
  const f16* arow = Ahat + m * KLDS + 16 * half;
  for (int kb = 0; kb < KAUG / 32; ++kb) {
    int k0 = kb * 32;
    v8h lo = *(const v8h*)(arow + k0);        // K = k0+16h .. +7
    v8h hi = *(const v8h*)(arow + k0 + 8);    // K = k0+16h+8 .. +7
    v16h a = __builtin_shufflevector(lo, hi, 0,1,2,3,4,5,6,7,8,9,10,11,12,13,14,15);
#pragma unroll
    for (int j = 0; j < 5; ++j) {
      int e = ebase + j * 16 + m;             // B frag: lane n=m, K run of 16
      v16h bf = *(const v16h*)(WgT + (size_t)e * KAUG + k0 + 16 * half);
      acc[j] = __builtin_amdgcn_wmma_f32_16x16x32_f16(
          false, a, false, bf, (short)0, acc[j], false, false);
    }
  }

  // Epilogue: pre = acc + s_comb[e]; psum[r] += tanh(pre)*W_ee[e]
  // C layout: lane's rows M = r + 8*half, column N = m.
  float psum[8];
#pragma unroll
  for (int r = 0; r < 8; ++r) psum[r] = 0.f;
#pragma unroll
  for (int j = 0; j < 5; ++j) {
    int e = ebase + j * 16 + m;
    float sc = scomb[b * Eh + e];
    float we = W_ee[e];
#pragma unroll
    for (int r = 0; r < 8; ++r)
      psum[r] += tanhf(acc[j][r] + sc) * we;
  }
  // reduce across the 16 lanes of each half (xor<16 stays inside the half)
  for (int off = 1; off < 16; off <<= 1) {
#pragma unroll
    for (int r = 0; r < 8; ++r) psum[r] += __shfl_xor(psum[r], off, 32);
  }
  if (m == 0) {
    int base = half * 8;
#pragma unroll
    for (int r = 0; r < 8; ++r) part[w][base + r] = psum[r];
  }
  __syncthreads();
  if (tid < 16) {
    float sum = 0.f;
#pragma unroll
    for (int ww = 0; ww < 8; ++ww) sum += part[ww][tid];
    sum += b_ee[0];
    int t = t0 + tid;
    e_ws[b * Tt + t] = mask[b * Tt + t] ? sum : -1e9f;
  }
}

// ---------------- softmax over T per batch ----------------
__global__ __launch_bounds__(256) void k_softmax(const float* __restrict__ e_ws,
                                                 float* __restrict__ alpha_out) {
  __shared__ float red[256];
  int b = blockIdx.x, tid = threadIdx.x;
  const float* e = e_ws + b * Tt;
  float m = -INFINITY;
  for (int t = tid; t < Tt; t += 256) m = fmaxf(m, e[t]);
  red[tid] = m; __syncthreads();
  for (int s = 128; s > 0; s >>= 1) {
    if (tid < s) red[tid] = fmaxf(red[tid], red[tid + s]);
    __syncthreads();
  }
  m = red[0]; __syncthreads();
  float sum = 0.f;
  for (int t = tid; t < Tt; t += 256) sum += expf(e[t] - m);
  red[tid] = sum; __syncthreads();
  for (int s = 128; s > 0; s >>= 1) {
    if (tid < s) red[tid] += red[tid + s];
    __syncthreads();
  }
  float inv = 1.f / red[0];
  for (int t = tid; t < Tt; t += 256)
    alpha_out[b * Tt + t] = expf(e[t] - m) * inv;
}

// ---------------- g = alpha^T h, chunked over t for parallelism ----------------
#define TCH 10
__global__ __launch_bounds__(320) void k_gpart(const float* __restrict__ h,
                                               const float* __restrict__ alpha,
                                               float* __restrict__ gpart) {
  int idx = blockIdx.x;           // grid = Bb * 2 * TCH
  int chunk = idx % TCH;
  int eh = (idx / TCH) % 2;
  int b = idx / (2 * TCH);
  int e = eh * 320 + threadIdx.x;
  const float* ab = alpha + b * Tt;
  const float* hb = h + (size_t)b * Tt * Eh + e;
  float acc = 0.f;
  int tb = chunk * (Tt / TCH), te = tb + (Tt / TCH);
  for (int t = tb; t < te; ++t) acc = fmaf(ab[t], hb[(size_t)t * Eh], acc);
  gpart[((size_t)chunk * Bb + b) * Eh + e] = acc;
}

__global__ void k_greduce(const float* __restrict__ gpart, float* __restrict__ g) {
  int idx = blockIdx.x * blockDim.x + threadIdx.x;
  if (idx >= Bb * Eh) return;
  float s = 0.f;
#pragma unroll
  for (int c = 0; c < TCH; ++c) s += gpart[(size_t)c * Bb * Eh + idx];
  g[idx] = s;
}

extern "C" void kernel_launch(void* const* d_in, const int* in_sizes, int n_in,
                              void* d_out, int out_size, void* d_ws, size_t ws_size,
                              hipStream_t stream) {
  const float* s      = (const float*)d_in[0];
  const float* h      = (const float*)d_in[1];
  const float* alpha  = (const float*)d_in[2];
  const unsigned char* mask = (const unsigned char*)d_in[3];
  const float* W_se   = (const float*)d_in[4];
  const float* b_se   = (const float*)d_in[5];
  const float* W_he   = (const float*)d_in[6];
  const float* b_he   = (const float*)d_in[7];
  const float* W_fe   = (const float*)d_in[8];
  const float* b_fe   = (const float*)d_in[9];
  const float* W_ee   = (const float*)d_in[10];
  const float* b_ee   = (const float*)d_in[11];
  const float* conv_w = (const float*)d_in[12];

  // workspace layout (base is 256B-aligned from hipMalloc)
  char* ws = (char*)d_ws;
  f16*   WgT   = (f16*)ws;                                  // 640*672*2 = 860160 B
  float* scomb = (float*)(ws + (size_t)Eh * KAUG * 2);      // 20480 f
  float* f_ws  = scomb + Bb * Eh;                           // 640000 f
  float* e_ws  = f_ws + (size_t)Bb * Tt * NFk;              // 64000 f
  float* gpart = e_ws + Bb * Tt;                            // 10*20480 f

  float* g_out     = (float*)d_out;       // [B,E]
  float* alpha_out = g_out + Bb * Eh;     // [B,T]

  k_scomb  <<<(Bb * Eh + 255) / 256, 256, 0, stream>>>(s, W_se, b_se, b_he, b_fe, scomb);
  k_wgt    <<<(Eh * KAUG + 255) / 256, 256, 0, stream>>>(W_he, W_fe, WgT);
  k_conv   <<<(Bb * Tt + 255) / 256, 256, 0, stream>>>(alpha, conv_w, f_ws);
  k_energy <<<Bb * (Tt / 16), 256, 0, stream>>>(h, f_ws, scomb, WgT, W_ee, b_ee, mask, e_ws);
  k_softmax<<<Bb, 256, 0, stream>>>(e_ws, alpha_out);
  k_gpart  <<<Bb * 2 * TCH, 320, 0, stream>>>(h, alpha_out, gpart);
  k_greduce<<<(Bb * Eh + 255) / 256, 256, 0, stream>>>(gpart, g_out);
}
